// FastRCNN_49160195670058
// MI455X (gfx1250) — matrix operand order repaired and checked by reading, not compiled, not run on hardware
//
#include <hip/hip_runtime.h>
#include <hip/hip_bf16.h>
#include <cstdint>
#include <cstddef>

// ---------------------------------------------------------------------------
// Types for CDNA5 WMMA (wave32, 16x16x32 f16 -> f32 accumulate)
// ---------------------------------------------------------------------------
typedef _Float16 half_t;
typedef __attribute__((ext_vector_type(16))) _Float16 v16h;
typedef __attribute__((ext_vector_type(8)))  _Float16 v8h;
typedef __attribute__((ext_vector_type(8)))  float    v8f;

__device__ __forceinline__ float silu_f(float v) { return v / (1.0f + __expf(-v)); }

#define MAX_KTAB 4608   // largest Cin*k*k in the net: 512*3*3
#define KCHUNK   128    // GEMM LDS B-panel K-chunk (divides 1024 and 6272)

// ---------------------------------------------------------------------------
// Pack fp32 matrix [rows][K] -> fp16 [Rpad][Kpad], zero-padded (weights).
// ---------------------------------------------------------------------------
__global__ void pack_f16_pad(const float* __restrict__ src, half_t* __restrict__ dst,
                             int rows, int K, int Kpad, int Rpad)
{
    long idx = (long)blockIdx.x * blockDim.x + threadIdx.x;
    long total = (long)Rpad * Kpad;
    if (idx >= total) return;
    int r = (int)(idx / Kpad);
    int k = (int)(idx - (long)r * Kpad);
    float v = (r < rows && k < K) ? src[(long)r * K + k] : 0.0f;
    dst[idx] = (half_t)v;
}

// Flat f32 -> f16 convert (activation shadow for kernel inputs).
__global__ void cvt_f16(const float* __restrict__ src, half_t* __restrict__ dst, long n)
{
    long idx = (long)blockIdx.x * blockDim.x + threadIdx.x;
    if (idx < n) dst[idx] = (half_t)src[idx];
}

// ---------------------------------------------------------------------------
// Implicit-GEMM convolution with WMMA, 4 N-tiles (64 output channels) / wave.
//   M = Nb*Ho*Wo, N = Cout (padded to 64), K = Cin*k*k (padded to 32)
// A: gathered from *fp16 shadow* activations (LDS decode table removes all
//    integer division; 1x1/s1/p0 fast path is pure strided f16 loads).
// B: pre-packed fp16 weights [Npad][Kpad], one 32B load per N-tile per K-step.
// Epilogue: bias + opt. SiLU + opt. f32 residual, dual f32+f16 store with
// channel-offset so concat targets are produced in place in both formats.
// ---------------------------------------------------------------------------
__global__ void __launch_bounds__(256)
conv_wmma(const half_t* __restrict__ x, const half_t* __restrict__ wB,
          const float* __restrict__ bias, const float* __restrict__ res,
          float* __restrict__ out, half_t* __restrict__ out16,
          int Nb, int Cin, int H, int W,
          int Cout, int Ho, int Wo,
          int k, int stride, int pad, int K, int Kpad,
          int outC, int coff, int act,
          int tilesM, int tilesN4)
{
    __shared__ int2 tbl[MAX_KTAB];
    const bool fast11 = (k == 1) && (stride == 1) && (pad == 0) && ((K & 31) == 0);

    if (!fast11) {                       // uniform across block
        int ksq = k * k;
        for (int idx = threadIdx.x; idx < Kpad; idx += 256) {
            int2 e;
            if (idx < K) {
                int ci = idx / ksq;      // divisions only here, once per layer
                int r2 = idx - ci * ksq;
                int ky = r2 / k, kx = r2 - ky * k;
                e.x = (ci * H + ky) * W + kx;
                e.y = (ky << 16) | kx;
            } else {                     // sentinel: always fails bounds check
                e.x = 0;
                e.y = (0x4000 << 16) | 0x4000;
            }
            tbl[idx] = e;
        }
        __syncthreads();
    }

    int wid = blockIdx.x * (blockDim.x >> 5) + (threadIdx.x >> 5);
    if (wid >= tilesM * tilesN4) return;         // uniform per wave
    int tm  = wid % tilesM;
    int tn4 = wid / tilesM;
    int lane = threadIdx.x & 31;
    int row  = lane & 15;
    int grp  = lane >> 4;

    int m    = tm * 16 + row;
    int HWo  = Ho * Wo;
    int nimg = m / HWo;
    int remm = m - nimg * HWo;
    int oy = remm / Wo, ox = remm - oy * Wo;
    int iy0 = oy * stride - pad;
    int ix0 = ox * stride - pad;
    int base = iy0 * W + ix0;
    const half_t* xin = x + (long)nimg * Cin * H * W;
    const int HWin = H * W;

    const half_t* bp = wB + (long)(tn4 * 64 + row) * Kpad + grp * 16;
    const long bstr = (long)16 * Kpad;

    v8f c[4] = {};
    const int kk_base = grp * 16;

    for (int k0 = 0; k0 < Kpad; k0 += 32) {
        __builtin_prefetch(bp + k0 + 128, 0, 0); // global_prefetch_b8
        v16h a;
        if (fast11) {
            const half_t* ap = xin + remm;       // pixel always in range
            #pragma unroll
            for (int i = 0; i < 16; ++i) {
                int off = kk_base + i;
                a[i] = ap[(long)(k0 + off) * HWin];
            }
        } else {
            #pragma unroll
            for (int i = 0; i < 16; ++i) {
                int off = kk_base + i;
                int2 e = tbl[k0 + off];          // ds_load, broadcast
                int ky = e.y >> 16, kx = e.y & 0xffff;
                int iy = iy0 + ky,  ix = ix0 + kx;
                half_t v = (half_t)0.0f;
                if ((unsigned)iy < (unsigned)H && (unsigned)ix < (unsigned)W)
                    v = xin[(long)e.x + base];
                a[i] = v;
            }
        }
        #pragma unroll
        for (int j = 0; j < 4; ++j) {
            v16h b = *(const v16h*)(bp + (long)j * bstr + k0);
            c[j] = __builtin_amdgcn_wmma_f32_16x16x32_f16(
                       false, a, false, b, (short)0, c[j], false, false);
        }
    }

    // ---- epilogue: division-free carry-increment over the 8 C rows ----
    int   nn[4]; float bb[4];
    #pragma unroll
    for (int j = 0; j < 4; ++j) {
        nn[j] = tn4 * 64 + j * 16 + row;
        bb[j] = (bias && nn[j] < Cout) ? bias[nn[j]] : 0.0f;
    }
    int mm0 = tm * 16 + grp * 8;
    int ni = mm0 / HWo;
    int rm = mm0 - ni * HWo;
    int yy = rm / Wo, xx = rm - yy * Wo;

    #pragma unroll
    for (int r = 0; r < 8; ++r) {
        long pbase = (((long)ni * outC + coff) * Ho + yy) * Wo + xx;
        long rbase = (((long)ni * Cout) * Ho + yy) * Wo + xx;
        #pragma unroll
        for (int j = 0; j < 4; ++j) {
            if (nn[j] < Cout) {
                float v = c[j][r] + bb[j];
                if (act) v = silu_f(v);
                if (res) v += res[rbase + (long)nn[j] * HWo];
                long pidx = pbase + (long)nn[j] * HWo;
                out[pidx] = v;
                if (out16) out16[pidx] = (half_t)v;
            }
        }
        if (++xx == Wo) { xx = 0; if (++yy == Ho) { yy = 0; ++ni; } }
    }
}

// ---------------------------------------------------------------------------
// Dense GEMM with WMMA. Block = 8 waves spanning 8 M-tiles, all sharing one
// 64-column N-stripe (blockIdx.y). The shared B panel is staged through LDS
// in double-buffered KCHUNK slices with GLOBAL_LOAD_ASYNC_TO_LDS_B128
// (ASYNCcnt + barrier), and B fragments are read back with ds_load_b128.
// Dual f32/f16 output (f16 feeds the next GEMM directly).
// ---------------------------------------------------------------------------
__global__ void __launch_bounds__(256)
gemm_wmma(const half_t* __restrict__ A, const half_t* __restrict__ Bt,
          const float* __restrict__ bias, float* __restrict__ outf,
          half_t* __restrict__ outh,
          int M, int N, int Kpad, int act, int tilesM)
{
    __shared__ half_t bsm[2][64 * KCHUNK];       // 2 x 16 KB

    int wave = threadIdx.x >> 5;
    int tm   = blockIdx.x * 8 + wave;
    int tn4  = blockIdx.y;
    bool active = (tm < tilesM);
    int tmc = active ? tm : tilesM - 1;          // clamp loads, skip stores
    int lane = threadIdx.x & 31;
    int row  = lane & 15;
    int grp  = lane >> 4;

    const half_t* ap   = A  + (long)(tmc * 16 + row) * Kpad + grp * 16;
    const half_t* brow = Bt + (long)tn4 * 64 * Kpad;

    // Stage 64 rows x KCHUNK halves (16 KB) of B into LDS buffer `buf`.
    auto stage = [&](int buf, int kbase) {
        for (int q = threadIdx.x; q < 64 * KCHUNK / 8; q += 256) {
            int rn = q >> 4;                     // KCHUNK/8 == 16 units/row
            int k8 = q & 15;
            const half_t* src = brow + (long)rn * Kpad + kbase + k8 * 8;
            uint32_t laddr = (uint32_t)(uintptr_t)&bsm[buf][rn * KCHUNK + k8 * 8];
            uint64_t gaddr = (uint64_t)(uintptr_t)src;
            asm volatile("global_load_async_to_lds_b128 %0, %1, off"
                         :: "v"(laddr), "v"(gaddr) : "memory");
        }
    };

    v8f c[4] = {};
    int nch = Kpad / KCHUNK;                     // Kpad % KCHUNK == 0 by host
    stage(0, 0);
    asm volatile("s_wait_asynccnt 0x0" ::: "memory");
    __syncthreads();

    for (int ch = 0; ch < nch; ++ch) {
        if (ch + 1 < nch) stage((ch + 1) & 1, (ch + 1) * KCHUNK);
        const half_t* bb = &bsm[ch & 1][0];
        int kg = ch * KCHUNK;
        for (int k0 = 0; k0 < KCHUNK; k0 += 32) {
            __builtin_prefetch(ap + kg + k0 + 256, 0, 0);
            v16h a;
            v8h t0 = *(const v8h*)(ap + kg + k0);
            v8h t1 = *(const v8h*)(ap + kg + k0 + 8);
            #pragma unroll
            for (int i = 0; i < 8; ++i) { a[i] = t0[i]; a[i + 8] = t1[i]; }
            #pragma unroll
            for (int j = 0; j < 4; ++j) {
                v16h b = *(const v16h*)(bb + (j * 16 + row) * KCHUNK + grp * 16 + k0);
                c[j] = __builtin_amdgcn_wmma_f32_16x16x32_f16(
                           false, a, false, b, (short)0, c[j], false, false);
            }
        }
        asm volatile("s_wait_asynccnt 0x0" ::: "memory");
        __syncthreads();                         // all waves done with buf
    }

    if (!active) return;
    #pragma unroll
    for (int j = 0; j < 4; ++j) {
        int nn = tn4 * 64 + j * 16 + row;
        if (nn >= N) continue;
        float bb2 = bias ? bias[nn] : 0.0f;
        #pragma unroll
        for (int r = 0; r < 8; ++r) {
            int mm = tm * 16 + grp * 8 + r;
            float v = c[j][r] + bb2;
            if (act) v = silu_f(v);
            if (outf) outf[(long)mm * N + nn] = v;
            if (outh) outh[(long)mm * N + nn] = (half_t)v;
        }
    }
}

// ---------------------------------------------------------------------------
// MaxPool 5x5 s1 p2; channel-offset in/out; dual f32+f16 store.
// ---------------------------------------------------------------------------
__global__ void maxpool5(const float* __restrict__ in, float* __restrict__ out,
                         half_t* __restrict__ out16,
                         int Nb, int C, int H, int W,
                         int inC, int ioff, int outC, int ooff)
{
    long idx = (long)blockIdx.x * blockDim.x + threadIdx.x;
    long total = (long)Nb * C * H * W;
    if (idx >= total) return;
    int xx = (int)(idx % W); long t = idx / W;
    int yy = (int)(t % H);   t /= H;
    int cc = (int)(t % C);
    int nn = (int)(t / C);
    const float* p = in + ((long)nn * inC + ioff + cc) * H * W;
    float m = -3.402823466e38f;
    for (int dy = -2; dy <= 2; ++dy) {
        int iy = yy + dy;
        if ((unsigned)iy >= (unsigned)H) continue;
        for (int dx = -2; dx <= 2; ++dx) {
            int ix = xx + dx;
            if ((unsigned)ix >= (unsigned)W) continue;
            m = fmaxf(m, p[(long)iy * W + ix]);
        }
    }
    long oidx = (((long)nn * outC + ooff + cc) * H + yy) * W + xx;
    out[oidx] = m;
    if (out16) out16[oidx] = (half_t)m;
}

// ---------------------------------------------------------------------------
// out = upsample2x_nearest(a) + b; dual f32+f16 store.
// ---------------------------------------------------------------------------
__global__ void up2_add(const float* __restrict__ a, const float* __restrict__ b,
                        float* __restrict__ out, half_t* __restrict__ out16,
                        int Nb, int C, int Ho, int Wo)
{
    long idx = (long)blockIdx.x * blockDim.x + threadIdx.x;
    long total = (long)Nb * C * Ho * Wo;
    if (idx >= total) return;
    int xx = (int)(idx % Wo); long t = idx / Wo;
    int yy = (int)(t % Ho);   t /= Ho;
    int cc = (int)(t % C);
    int nn = (int)(t / C);
    int Hi = Ho >> 1, Wi = Wo >> 1;
    float v = a[(((long)nn * C + cc) * Hi + (yy >> 1)) * Wi + (xx >> 1)] + b[idx];
    out[idx] = v;
    out16[idx] = (half_t)v;
}

// ===========================================================================
// Host-side wiring
// ===========================================================================
static inline int rup(int x, int m) { return (x + m - 1) / m * m; }
static inline long cdiv(long a, long b) { return (a + b - 1) / b; }

struct Bump {
    char* base; size_t off;
    void* alloc(size_t bytes) {
        size_t a = (off + 255) & ~(size_t)255;
        off = a + bytes;
        return base + a;
    }
};

struct Pp { const float* w; const float* b; };
struct Cur {
    void* const* in; int i;
    Pp cv() { Pp p; p.w = (const float*)in[i++]; p.b = (const float*)in[i++]; return p; }
};
struct Tens { float* f; half_t* h; };   // f32 tensor + f16 shadow

static Tens allocT(Bump& ws, size_t n)
{
    Tens t;
    t.f = (float*)ws.alloc(n * sizeof(float));
    t.h = (half_t*)ws.alloc(n * sizeof(half_t));
    return t;
}

// out16 == nullptr means "f32-only output" (detector heads into d_out).
static void conv_l(hipStream_t st, Bump& ws, Pp p,
                   const half_t* x, float* out, half_t* out16,
                   int Nb, int Cin, int H, int W, int Cout,
                   int k, int s, int pad,
                   int outC, int coff, int act, const float* res)
{
    int Ho = (H + 2 * pad - k) / s + 1;
    int Wo = (W + 2 * pad - k) / s + 1;
    int K = Cin * k * k, Kpad = rup(K, 32), Npad = rup(Cout, 64);
    half_t* wp = (half_t*)ws.alloc((size_t)Npad * Kpad * sizeof(half_t));
    pack_f16_pad<<<(int)cdiv((long)Npad * Kpad, 256), 256, 0, st>>>(p.w, wp, Cout, K, Kpad, Npad);
    int M = Nb * Ho * Wo;
    int tilesM = M / 16, tilesN4 = Npad / 64;
    int waves = tilesM * tilesN4;
    conv_wmma<<<(waves + 7) / 8, 256, 0, st>>>(x, wp, p.b, res, out, out16,
                                               Nb, Cin, H, W, Cout, Ho, Wo,
                                               k, s, pad, K, Kpad, outC, coff, act,
                                               tilesM, tilesN4);
}

// CSPF: c1 -> cat[0:h]; residual chain ping-pong (residual fused in epilogue);
// last residual conv writes cat[h:2h]; then c3 over the concat.
static Tens csp_l(hipStream_t st, Bump& ws, Cur& c, const Tens x,
                  int Nb, int cin, int cout, int H, int W)
{
    int h = cout / 2;
    Pp c1 = c.cv(), c2 = c.cv(), r0 = c.cv(), r1 = c.cv(), r2 = c.cv(), c3 = c.cv();
    size_t hw = (size_t)Nb * H * W;
    Tens cat = allocT(ws, hw * cout);
    Tens b0  = allocT(ws, hw * h);
    Tens b1  = allocT(ws, hw * h);
    conv_l(st, ws, c1, x.h,  cat.f, cat.h, Nb, cin, H, W, h, 1, 1, 0, cout, 0, 1, nullptr);
    conv_l(st, ws, c2, x.h,  b0.f,  b0.h,  Nb, cin, H, W, h, 1, 1, 0, h,    0, 1, nullptr);
    conv_l(st, ws, r0, b0.h, b1.f,  b1.h,  Nb, h,   H, W, h, 3, 1, 1, h,    0, 1, b0.f);
    conv_l(st, ws, r1, b1.h, b0.f,  b0.h,  Nb, h,   H, W, h, 3, 1, 1, h,    0, 1, b1.f);
    conv_l(st, ws, r2, b0.h, cat.f, cat.h, Nb, h,   H, W, h, 3, 1, 1, cout, h, 1, b0.f);
    Tens out = allocT(ws, hw * cout);
    conv_l(st, ws, c3, cat.h, out.f, out.h, Nb, cout, H, W, cout, 1, 1, 0, cout, 0, 1, nullptr);
    return out;
}

// SPPF: c1 -> cat[0:512], three chained maxpool5 into cat slots, c2 over 2048ch.
static Tens spp_l(hipStream_t st, Bump& ws, Cur& c, const Tens x,
                  int Nb, int H, int W)
{
    Pp c1 = c.cv(), c2 = c.cv();
    size_t hw = (size_t)Nb * H * W;
    Tens cat = allocT(ws, hw * 2048);
    conv_l(st, ws, c1, x.h, cat.f, cat.h, Nb, 1024, H, W, 512, 1, 1, 0, 2048, 0, 1, nullptr);
    long tot = (long)Nb * 512 * H * W;
    for (int i = 0; i < 3; ++i)
        maxpool5<<<(int)cdiv(tot, 256), 256, 0, st>>>(cat.f, cat.f, cat.h, Nb, 512, H, W,
                                                      2048, i * 512, 2048, (i + 1) * 512);
    Tens out = allocT(ws, hw * 1024);
    conv_l(st, ws, c2, cat.h, out.f, out.h, Nb, 2048, H, W, 1024, 1, 1, 0, 1024, 0, 1, nullptr);
    return out;
}

static Tens up_csp(hipStream_t st, Bump& ws, Cur& cur,
                   const Tens hi, const Tens lat, int Nb, int H, int W)
{
    Tens u = allocT(ws, (size_t)Nb * 128 * H * W);
    up2_add<<<(int)cdiv((long)Nb * 128 * H * W, 256), 256, 0, st>>>(hi.f, lat.f, u.f, u.h,
                                                                    Nb, 128, H, W);
    return csp_l(st, ws, cur, u, Nb, 128, 128, H, W);
}

extern "C" void kernel_launch(void* const* d_in, const int* in_sizes, int n_in,
                              void* d_out, int out_size, void* d_ws, size_t ws_size,
                              hipStream_t stream)
{
    (void)in_sizes; (void)n_in; (void)out_size; (void)ws_size;
    Bump ws{(char*)d_ws, 0};
    Cur cur{d_in, 0};
    const float* images = (const float*)d_in[cur.i++];
    const float* roi    = (const float*)d_in[cur.i++];
    const int Nb = 2;

    auto ahp = [&](size_t n) { return (half_t*)ws.alloc(n * sizeof(half_t)); };
    auto cvt = [&](const float* s, long n) {
        half_t* d = ahp((size_t)n);
        cvt_f16<<<(int)cdiv(n, 256), 256, 0, stream>>>(s, d, n);
        return d;
    };

    // ---- DarkNet backbone ----
    half_t* img16 = cvt(images, (long)Nb * 3 * 640 * 640);
    Pp s0 = cur.cv(), s1 = cur.cv(), s2 = cur.cv();
    Tens x0 = allocT(ws, (size_t)Nb * 64 * 320 * 320);
    conv_l(stream, ws, s0, img16, x0.f, x0.h, Nb, 3, 640, 640, 64, 6, 2, 2, 64, 0, 1, nullptr);
    Tens x1 = allocT(ws, (size_t)Nb * 64 * 320 * 320);
    conv_l(stream, ws, s1, x0.h, x1.f, x1.h, Nb, 64, 320, 320, 64, 3, 1, 1, 64, 0, 1, x0.f);
    Tens x2 = allocT(ws, (size_t)Nb * 128 * 160 * 160);
    conv_l(stream, ws, s2, x1.h, x2.f, x2.h, Nb, 64, 320, 320, 128, 3, 2, 1, 128, 0, 1, nullptr);
    Tens p2 = csp_l(stream, ws, cur, x2, Nb, 128, 128, 160, 160);

    Pp d3 = cur.cv();
    Tens x3 = allocT(ws, (size_t)Nb * 256 * 80 * 80);
    conv_l(stream, ws, d3, p2.h, x3.f, x3.h, Nb, 128, 160, 160, 256, 3, 2, 1, 256, 0, 1, nullptr);
    Tens t  = csp_l(stream, ws, cur, x3, Nb, 256, 256, 80, 80);
    Tens p3 = csp_l(stream, ws, cur, t,  Nb, 256, 256, 80, 80);

    Pp d4 = cur.cv();
    Tens x4 = allocT(ws, (size_t)Nb * 512 * 40 * 40);
    conv_l(stream, ws, d4, p3.h, x4.f, x4.h, Nb, 256, 80, 80, 512, 3, 2, 1, 512, 0, 1, nullptr);
    t = csp_l(stream, ws, cur, x4, Nb, 512, 512, 40, 40);
    t = csp_l(stream, ws, cur, t,  Nb, 512, 512, 40, 40);
    Tens p4 = csp_l(stream, ws, cur, t, Nb, 512, 512, 40, 40);

    Pp d5 = cur.cv();
    Tens x5 = allocT(ws, (size_t)Nb * 1024 * 20 * 20);
    conv_l(stream, ws, d5, p4.h, x5.f, x5.h, Nb, 512, 40, 40, 1024, 3, 2, 1, 1024, 0, 1, nullptr);
    t = csp_l(stream, ws, cur, x5, Nb, 1024, 1024, 20, 20);
    Tens p5 = spp_l(stream, ws, cur, t, Nb, 20, 20);

    // ---- lateral 1x1 convs to 128ch ----
    Pp la = cur.cv(), lb = cur.cv(), lc = cur.cv(), ld = cur.cv();
    Tens l2 = allocT(ws, (size_t)Nb * 128 * 160 * 160);
    conv_l(stream, ws, la, p2.h, l2.f, l2.h, Nb, 128, 160, 160, 128, 1, 1, 0, 128, 0, 1, nullptr);
    Tens l3 = allocT(ws, (size_t)Nb * 128 * 80 * 80);
    conv_l(stream, ws, lb, p3.h, l3.f, l3.h, Nb, 256, 80, 80, 128, 1, 1, 0, 128, 0, 1, nullptr);
    Tens l4 = allocT(ws, (size_t)Nb * 128 * 40 * 40);
    conv_l(stream, ws, lc, p4.h, l4.f, l4.h, Nb, 512, 40, 40, 128, 1, 1, 0, 128, 0, 1, nullptr);
    Tens l5 = allocT(ws, (size_t)Nb * 128 * 20 * 20);
    conv_l(stream, ws, ld, p5.h, l5.f, l5.h, Nb, 1024, 20, 20, 128, 1, 1, 0, 128, 0, 1, nullptr);

    // ---- DarkFPN ----
    Tens h0 = up_csp(stream, ws, cur, l5, l4, Nb, 40, 40);
    Tens h1 = up_csp(stream, ws, cur, h0, l3, Nb, 80, 80);
    Tens h2 = up_csp(stream, ws, cur, h1, l2, Nb, 160, 160);

    Pp h3p = cur.cv();
    Tens dn0 = allocT(ws, (size_t)Nb * 128 * 80 * 80);
    conv_l(stream, ws, h3p, h2.h, dn0.f, dn0.h, Nb, 128, 160, 160, 128, 3, 2, 1, 128, 0, 1, h1.f);
    Tens h4 = csp_l(stream, ws, cur, dn0, Nb, 128, 128, 80, 80);

    Pp h5p = cur.cv();
    Tens dn1 = allocT(ws, (size_t)Nb * 128 * 40 * 40);
    conv_l(stream, ws, h5p, h4.h, dn1.f, dn1.h, Nb, 128, 80, 80, 128, 3, 2, 1, 128, 0, 1, h0.f);
    Tens h6 = csp_l(stream, ws, cur, dn1, Nb, 128, 128, 40, 40);

    Pp h7p = cur.cv();
    Tens dn2 = allocT(ws, (size_t)Nb * 128 * 20 * 20);
    conv_l(stream, ws, h7p, h6.h, dn2.f, dn2.h, Nb, 128, 40, 40, 128, 3, 2, 1, 128, 0, 1, l5.f);
    Tens h8 = csp_l(stream, ws, cur, dn2, Nb, 128, 128, 20, 20);

    // ---- RPN heads (1x1, no activation) written straight to d_out ----
    Pp pcls = cur.cv(), preg = cur.cv();
    float* o = (float*)d_out;
    size_t off = 0;
    const half_t* feats[4] = {h2.h, h4.h, h6.h, h8.h};
    const int     hsz[4]   = {160, 80, 40, 20};
    for (int i = 0; i < 4; ++i) {
        conv_l(stream, ws, pcls, feats[i], o + off, nullptr, Nb, 128, hsz[i], hsz[i], 9, 1, 1, 0, 9, 0, 0, nullptr);
        off += (size_t)Nb * 9 * hsz[i] * hsz[i];
    }
    for (int i = 0; i < 4; ++i) {
        conv_l(stream, ws, preg, feats[i], o + off, nullptr, Nb, 128, hsz[i], hsz[i], 36, 1, 1, 0, 36, 0, 0, nullptr);
        off += (size_t)Nb * 36 * hsz[i] * hsz[i];
    }

    // ---- Box head MLP (async-staged WMMA GEMMs, f16 chained) ----
    Pp f1 = cur.cv(), f2 = cur.cv(), wcls = cur.cv(), wreg = cur.cv();
    const int M = 2000;

    auto gemm = [&](const half_t* A, Pp wp, int m, int n, int k, int act,
                    float* outf, half_t* outh) {
        int Npad = rup(n, 64), Kpad = rup(k, 32);   // K % KCHUNK == 0 here
        half_t* Wp = ahp((size_t)Npad * Kpad);
        pack_f16_pad<<<(int)cdiv((long)Npad * Kpad, 256), 256, 0, stream>>>(wp.w, Wp, n, k, Kpad, Npad);
        int tilesM = m / 16, tilesN4 = Npad / 64;
        dim3 grid((unsigned)cdiv(tilesM, 8), (unsigned)tilesN4);
        gemm_wmma<<<grid, 256, 0, stream>>>(A, Wp, wp.b, outf, outh, m, n, Kpad, act, tilesM);
    };

    half_t* A0 = cvt(roi, (long)M * 6272);
    half_t* A1 = ahp((size_t)M * 1024);
    gemm(A0, f1, M, 1024, 6272, 1, nullptr, A1);
    half_t* A2 = ahp((size_t)M * 1024);
    gemm(A1, f2, M, 1024, 1024, 1, nullptr, A2);

    gemm(A2, wcls, M, 91, 1024, 0, o + off, nullptr);   off += (size_t)M * 91;
    gemm(A2, wreg, M, 364, 1024, 0, o + off, nullptr);
}
